// CpsTcnModel_59459527246324
// MI455X (gfx1250) — matrix operand order repaired and checked by reading, not compiled, not run on hardware
//
#include <hip/hip_runtime.h>
#include <hip/hip_bf16.h>

// ---------------- problem constants ----------------
constexpr int B_    = 32;
constexpr int L_    = 1024;
constexpr int C_    = 512;
constexpr int K_    = 2;
constexpr int NLEV_ = 4;
constexpr int OUT_  = 11;

// ---------------- WMMA vector types ----------------
typedef __attribute__((ext_vector_type(16))) __bf16 v16bf;
typedef __attribute__((ext_vector_type(8)))  __bf16 v8bf;
typedef __attribute__((ext_vector_type(8)))  float  v8f;

// bf16 helpers (storage = raw u16, round-to-nearest-even on pack)
__device__ __forceinline__ unsigned short f2bf(float f) {
    unsigned int u = __float_as_uint(f);
    u = (u + 0x7FFFu + ((u >> 16) & 1u)) >> 16;
    return (unsigned short)u;
}
__device__ __forceinline__ float bf2f(unsigned short h) {
    return __uint_as_float(((unsigned int)h) << 16);
}

// ---------------------------------------------------------------------------
// 1) Convert fp32 weights [NLEV,C,C,K] (x2 tensors) into bf16 laid out as
//    [tensor][lev][tap][co][ci]  (ci contiguous -> A-operand rows contiguous)
// ---------------------------------------------------------------------------
__global__ __launch_bounds__(256) void tcn_wconv_kernel(
    const float* __restrict__ w1, const float* __restrict__ w2,
    unsigned short* __restrict__ dst)
{
    const int per = NLEV_ * C_ * C_ * K_;                  // elems per tensor
    int e = blockIdx.x * 256 + threadIdx.x;                // over 2*per
    int t = e / per;
    int r = e - t * per;
    const float* src = t ? w2 : w1;
    int lev = r / (C_ * C_ * K_);
    int r2  = r % (C_ * C_ * K_);
    int co  = r2 / (C_ * K_);
    int r3  = r2 % (C_ * K_);
    int ci  = r3 / K_;
    int k   = r3 % K_;
    size_t d = (size_t)t * per + (((size_t)(lev * 2 + k) * C_ + co) * C_ + ci);
    dst[d] = f2bf(src[r]);
}

// ---------------------------------------------------------------------------
// 2) Embedding gather: h0[b,l,c] = bf16(emb[x[b,l], c]), layout [B,L,C]
// ---------------------------------------------------------------------------
__global__ __launch_bounds__(256) void tcn_embed_kernel(
    const int* __restrict__ x, const float* __restrict__ emb,
    unsigned short* __restrict__ h)
{
    int t   = blockIdx.x * 256 + threadIdx.x;              // over B*L*(C/4)
    int c4  = (t % (C_ / 4)) * 4;
    int bl  = t / (C_ / 4);
    int tok = x[bl];
    const float* e = emb + (size_t)tok * C_ + c4;
    unsigned short* o = h + (size_t)bl * C_ + c4;
    o[0] = f2bf(e[0]); o[1] = f2bf(e[1]); o[2] = f2bf(e[2]); o[3] = f2bf(e[3]);
}

// ---------------------------------------------------------------------------
// 3) Causal dilated conv (K=2) as dual-tap WMMA GEMM + bias + ReLU
//    (+ optional residual + ReLU).  hin/hout: bf16 [B,L,C].
//    wk: bf16 [2][C][C]  (tap-major, row co, contiguous ci)
//
//    Block = 8 waves: 32 l-columns x 128 co.  One LDS tile of 40 rows
//    (seq rows l0-8 .. l0+31; front pad covers dilation <= 8) is staged once
//    per block with GLOBAL_LOAD_ASYNC_TO_LDS_B128 (ASYNCcnt).  Each wave then
//    computes 16co x 32l with two f32 accumulators: every A (weight) fragment
//    from L2 feeds two v_wmma_f32_16x16x32_bf16 -> 64 WMMA per wave (fully
//    unrolled K loop), and L2 weight traffic is halved versus 16-wide N.
// ---------------------------------------------------------------------------
__global__ __launch_bounds__(256) void tcn_conv_wmma_kernel(
    const unsigned short* __restrict__ hin,
    const unsigned short* __restrict__ wk,
    const float*          __restrict__ bias,
    const unsigned short* __restrict__ res,    // nullptr => no residual
    unsigned short*       __restrict__ hout,
    int dil)
{
    constexpr int PAD  = 520;      // row stride (elems): 260 dwords % 64 = 4
    constexpr int ROWS = 40;       // 8 front-pad rows + 32 data rows
    __shared__ unsigned short smem[ROWS * PAD];

    const int tid  = threadIdx.x;
    const int lane = tid & 31;
    const int wave = tid >> 5;

    const int lt  = blockIdx.x >> 2;           // 32-wide (b,l) tile: 1024 total
    const int cb  = blockIdx.x & 3;            // co block (128 co each)
    const int b   = lt >> 5;                   // L/32 = 32 l-tiles per batch
    const int l0  = (lt & 31) << 5;
    const int co0 = (cb * 8 + wave) << 4;

    const size_t seqbase = (size_t)b * L_ * C_;
    const unsigned short* gbase = hin + seqbase;

    // ---- causal zero-pad: 8 front rows (seq rows < 0) at sequence start
    if (l0 == 0) {
        for (int i = tid; i < 8 * C_; i += 256) {
            int r = i >> 9, c = i & (C_ - 1);
            smem[r * PAD + c] = 0;
        }
    }
    // ---- async global->LDS stage: 40 rows x 64 16B-chunks = 2560 chunks
#pragma unroll
    for (int t = 0; t < 10; ++t) {
        const int chunk = tid + t * 256;
        const int row   = chunk >> 6;          // LDS row 0..39
        const int c8    = (chunk & 63) << 3;   // element column (x8)
        const int sr    = l0 - 8 + row;        // sequence row
        if (sr >= 0) {
            unsigned ldsb = (unsigned)((row * PAD + c8) * 2);
            unsigned voff = (unsigned)((sr * C_ + c8) * 2);
            asm volatile("global_load_async_to_lds_b128 %0, %1, %2"
                         :: "v"(ldsb), "v"(voff),
                            "s"((unsigned long long)(uintptr_t)gbase)
                         : "memory");
        }
    }
    asm volatile("s_wait_asynccnt 0x0" ::: "memory");
    __syncthreads();

    // ---- WMMA main loop: 16co x 32l per wave (2 accumulators) ----
    const int hi = lane >> 4;                  // lane half (K-half select)
    const int lm = lane & 15;
    const int coA = co0 + lm;                  // A row for this lane

    v8f acc0, acc1;                            // C/D: VGPR j -> M = j + 8*hi
#pragma unroll
    for (int j = 0; j < 8; ++j) { acc0[j] = bias[co0 + j + hi * 8]; acc1[j] = acc0[j]; }

    // hoisted base pointers: K-loop offsets become compile-time immediates
    const unsigned short* W0 = wk + (size_t)coA * C_ + hi * 16;       // tap 0
    const unsigned short* W1 = W0 + C_ * C_;                          // tap 1
    const unsigned short* cur0 = &smem[(8 + lm)      * PAD] + hi * 16;
    const unsigned short* cur1 = &smem[(8 + 16 + lm) * PAD] + hi * 16;
    const unsigned short* sh0  = cur0 - dil * PAD;
    const unsigned short* sh1  = cur1 - dil * PAD;

#pragma unroll
    for (int kk = 0; kk < C_; kk += 32) {
        union Ubf { v16bf v; v8bf h2[2]; };
        Ubf a0, a1, bb;
        a0.h2[0] = *(const v8bf*)(W0 + kk);
        a0.h2[1] = *(const v8bf*)(W0 + kk + 8);
        a1.h2[0] = *(const v8bf*)(W1 + kk);
        a1.h2[1] = *(const v8bf*)(W1 + kk + 8);

        // subtile 0 (columns l0 .. l0+15)
        bb.h2[0] = *(const v8bf*)(sh0 + kk);
        bb.h2[1] = *(const v8bf*)(sh0 + kk + 8);
        acc0 = __builtin_amdgcn_wmma_f32_16x16x32_bf16(
                   false, a0.v, false, bb.v, (short)0, acc0, false, false);
        bb.h2[0] = *(const v8bf*)(cur0 + kk);
        bb.h2[1] = *(const v8bf*)(cur0 + kk + 8);
        acc0 = __builtin_amdgcn_wmma_f32_16x16x32_bf16(
                   false, a1.v, false, bb.v, (short)0, acc0, false, false);
        // subtile 1 (columns l0+16 .. l0+31) -- reuses a0/a1 fragments
        bb.h2[0] = *(const v8bf*)(sh1 + kk);
        bb.h2[1] = *(const v8bf*)(sh1 + kk + 8);
        acc1 = __builtin_amdgcn_wmma_f32_16x16x32_bf16(
                   false, a0.v, false, bb.v, (short)0, acc1, false, false);
        bb.h2[0] = *(const v8bf*)(cur1 + kk);
        bb.h2[1] = *(const v8bf*)(cur1 + kk + 8);
        acc1 = __builtin_amdgcn_wmma_f32_16x16x32_bf16(
                   false, a1.v, false, bb.v, (short)0, acc1, false, false);
    }

    // ---- epilogue: ReLU (+ residual + ReLU), store bf16 ----
#pragma unroll
    for (int st = 0; st < 2; ++st) {
        const v8f& acc = st ? acc1 : acc0;
        const size_t rowD = seqbase + (size_t)(l0 + st * 16 + lm) * C_;
#pragma unroll
        for (int j = 0; j < 8; ++j) {
            const int coj = co0 + j + hi * 8;
            const size_t oi = rowD + coj;
            float f = fmaxf(acc[j], 0.0f);
            if (res) f = fmaxf(f + bf2f(res[oi]), 0.0f);
            hout[oi] = f2bf(f);
        }
    }
}

// ---------------------------------------------------------------------------
// 4) Decoder: y[b,l,o] = mask[b,l] ? (h[b,l,:] . dec_w[o,:] + dec_b[o]) : 0
//    OUT=11 is tiny -> plain VALU dot products (pad o to 16 per (b,l)).
// ---------------------------------------------------------------------------
__global__ __launch_bounds__(256) void tcn_decode_kernel(
    const unsigned short* __restrict__ h,
    const float* __restrict__ dec_w, const float* __restrict__ dec_b,
    const unsigned char* __restrict__ mask,
    float* __restrict__ out)
{
    int idx = blockIdx.x * 256 + threadIdx.x;  // over B*L*16
    int o  = idx & 15;
    int bl = idx >> 4;
    if (bl >= B_ * L_ || o >= OUT_) return;
    const unsigned short* row = h + (size_t)bl * C_;
    const float* wr = dec_w + (size_t)o * C_;
    float s = 0.0f;
#pragma unroll 4
    for (int c = 0; c < C_; ++c) s = fmaf(bf2f(row[c]), wr[c], s);
    s += dec_b[o];
    out[(size_t)bl * OUT_ + o] = mask[bl] ? s : 0.0f;
}

// ---------------------------------------------------------------------------
extern "C" void kernel_launch(void* const* d_in, const int* in_sizes, int n_in,
                              void* d_out, int out_size, void* d_ws, size_t ws_size,
                              hipStream_t stream)
{
    const int*           x     = (const int*)          d_in[0];
    const unsigned char* mask  = (const unsigned char*)d_in[1];
    const float*         emb   = (const float*)        d_in[2];
    const float*         w1    = (const float*)        d_in[3];
    const float*         b1    = (const float*)        d_in[4];
    const float*         w2    = (const float*)        d_in[5];
    const float*         b2    = (const float*)        d_in[6];
    const float*         dec_w = (const float*)        d_in[7];
    const float*         dec_b = (const float*)        d_in[8];
    float*               out   = (float*)              d_out;

    // workspace: bf16 weights (8 MB) + three bf16 activation buffers (3x32 MB)
    unsigned short* wbf = (unsigned short*)d_ws;
    const size_t wElems = (size_t)2 * NLEV_ * 2 * C_ * C_;
    unsigned short* h0 = wbf + wElems;
    unsigned short* h1 = h0 + (size_t)B_ * L_ * C_;
    unsigned short* h2 = h1 + (size_t)B_ * L_ * C_;

    // 1) weight conversion: 2 * NLEV*C*C*K elements
    {
        int n = 2 * NLEV_ * C_ * C_ * K_;
        tcn_wconv_kernel<<<n / 256, 256, 0, stream>>>(w1, w2, wbf);
    }
    // 2) embedding gather
    {
        int n = B_ * L_ * (C_ / 4);
        tcn_embed_kernel<<<n / 256, 256, 0, stream>>>(x, emb, h0);
    }
    // 3) TCN levels
    const int blocks = (B_ * L_ / 32) * (C_ / 128);  // 1024 l-tiles x 4 co-blocks
    unsigned short *hin = h0, *ha = h1, *hb = h2;
    for (int lev = 0; lev < NLEV_; ++lev) {
        const int d = 1 << lev;
        const unsigned short* wl1 = wbf + (size_t)lev * 2 * C_ * C_;
        const unsigned short* wl2 = wbf + (size_t)NLEV_ * 2 * C_ * C_
                                        + (size_t)lev * 2 * C_ * C_;
        tcn_conv_wmma_kernel<<<blocks, 256, 0, stream>>>(
            hin, wl1, b1 + lev * C_, nullptr, ha, d);
        tcn_conv_wmma_kernel<<<blocks, 256, 0, stream>>>(
            ha, wl2, b2 + lev * C_, hin, hb, d);
        unsigned short* t = hin; hin = hb; hb = t;   // rotate (ha stays scratch)
    }
    // 4) decoder + mask
    {
        int n = B_ * L_ * 16;
        tcn_decode_kernel<<<n / 256, 256, 0, stream>>>(hin, dec_w, dec_b, mask, out);
    }
}
